// NSubComplex_8667244003693
// MI455X (gfx1250) — matrix operand order repaired and checked by reading, not compiled, not run on hardware
//
#include <hip/hip_runtime.h>
#include <hip/hip_bf16.h>

typedef _Float16 half16 __attribute__((ext_vector_type(16)));
typedef float    v8f    __attribute__((ext_vector_type(8)));
typedef unsigned int u32x4 __attribute__((ext_vector_type(4)));

#define NPTS    2000000
#define NTILES  (NPTS / 32)     // 62500, each wave-iteration covers 32 points
#define FEATK   128             // 105 features zero-padded to 4 * 32
#define HID     64

union FragU {
    struct { u32x4 lo, hi; } q;
    half16 h;
};

__device__ __forceinline__ v8f wmma_f16(half16 a, half16 b, v8f c) {
    return __builtin_amdgcn_wmma_f32_16x16x32_f16(
        /*neg_a=*/false, a, /*neg_b=*/false, b,
        /*c_mod=*/(short)0, c, /*reuse_a=*/false, /*reuse_b=*/false);
}

// A-fragment (16xK row-major in LDS): lane ln holds row M=ln.
// Lanes 0-15 take K = kt*32 + 0..15; lanes 16-31 take +16.
// row stride in halfs must be a multiple of 8 (16B).
__device__ __forceinline__ half16 load_A(const _Float16* row, int kt, int hs) {
    const u32x4* p = (const u32x4*)row;      // 8 halfs per u32x4
    FragU f;
    f.q.lo = p[kt * 4 + 2 * hs];
    f.q.hi = p[kt * 4 + 2 * hs + 1];
    return f.h;
}

// B-fragment: weights stored [n][k] in LDS; lane ln = column, hs selects the
// contiguous 16-K half (lanes 0-15: K = kt*32 + 0..15, lanes 16-31: +16).
__device__ __forceinline__ half16 load_B(const _Float16* p16) {
    const u32x4* p = (const u32x4*)p16;
    FragU f;
    f.q.lo = p[0];
    f.q.hi = p[1];
    return f.h;
}

__global__ void __launch_bounds__(64)
neural_points_kernel(const float* __restrict__ bases,
                     const float* __restrict__ normals,
                     const float* __restrict__ encodings,
                     const float* __restrict__ W1, const float* __restrict__ b1,
                     const float* __restrict__ W2, const float* __restrict__ b2,
                     const float* __restrict__ W3, const float* __restrict__ b3,
                     float* __restrict__ out) {
    // LDS: weights (block-shared, f16, [n][k]) + per-wave feature/activation tiles
    __shared__ __align__(16) _Float16 sW1[HID * FEATK];   // 16 KB
    __shared__ __align__(16) _Float16 sW2[HID * HID];     //  8 KB
    __shared__ __align__(16) _Float16 sW3[16 * HID];      //  2 KB  (N padded 3->16)
    __shared__ float sB1[HID], sB2[HID], sB3[16];
    __shared__ __align__(16) _Float16 sFeat[2][32 * FEATK]; // 16 KB (2 waves x 32 rows)
    __shared__ __align__(16) _Float16 sAct[2][16 * HID];    //  4 KB

    const int tid  = threadIdx.x;
    const int lane = tid & 31;
    const int w    = tid >> 5;        // wave in block (0..1)
    const int ln   = lane & 15;       // column / row-within-tile
    const int hs   = lane >> 4;       // half-select for K striping

    // ---- Stage weights to LDS as f16, transposed to [n][k], zero-padded ----
    for (int i = tid; i < HID * FEATK; i += blockDim.x) {
        int n = i >> 7, k = i & (FEATK - 1);
        float v = (k < 105) ? W1[k * HID + n] : 0.0f;
        sW1[n * FEATK + k] = (_Float16)v;
    }
    for (int i = tid; i < HID * HID; i += blockDim.x) {
        int n = i >> 6, k = i & 63;
        sW2[n * HID + k] = (_Float16)W2[k * HID + n];
    }
    for (int i = tid; i < 16 * HID; i += blockDim.x) {
        int n = i >> 6, k = i & 63;
        float v = (n < 3) ? W3[k * 3 + n] : 0.0f;
        sW3[n * HID + k] = (_Float16)v;
    }
    if (tid < HID) { sB1[tid] = b1[tid]; sB2[tid] = b2[tid]; }
    if (tid < 16)  { sB3[tid] = (tid < 3) ? b3[tid] : 0.0f; }
    __syncthreads();

    const int wv = blockIdx.x * (blockDim.x >> 5) + w;
    const int nw = gridDim.x * (blockDim.x >> 5);

    for (int tile = wv; tile < NTILES; tile += nw) {
        const int base = tile * 32;

        // ---- Feature encoding: lane handles point base+lane (32 rows) ----
        {
            const int p = base + lane;
            _Float16* fr = &sFeat[w][lane * FEATK];
            const float bx = bases[p * 3 + 0];
            const float by = bases[p * 3 + 1];
            const float bz = bases[p * 3 + 2];
            const float n0 = normals[p * 2 + 0];
            const float n1 = normals[p * 2 + 1];
            // 20 learned encodings (p*20 floats => 16B aligned, load as float4)
            const float4* ep = (const float4*)(encodings + p * 20);
#pragma unroll
            for (int j = 0; j < 5; ++j) {
                float4 e = ep[j];
                fr[j * 4 + 0] = (_Float16)e.x;
                fr[j * 4 + 1] = (_Float16)e.y;
                fr[j * 4 + 2] = (_Float16)e.z;
                fr[j * 4 + 3] = (_Float16)e.w;
            }
            fr[20] = (_Float16)bx; fr[21] = (_Float16)by; fr[22] = (_Float16)bz;
            float sc = 2.0f;
#pragma unroll
            for (int i = 0; i < 8; ++i) {
                const int o = 23 + i * 6;
                fr[o + 0] = (_Float16)__sinf(sc * bx);
                fr[o + 1] = (_Float16)__sinf(sc * by);
                fr[o + 2] = (_Float16)__sinf(sc * bz);
                fr[o + 3] = (_Float16)__cosf(sc * bx);
                fr[o + 4] = (_Float16)__cosf(sc * by);
                fr[o + 5] = (_Float16)__cosf(sc * bz);
                sc *= 2.0f;
            }
            fr[71] = (_Float16)n0; fr[72] = (_Float16)n1;
#pragma unroll
            for (int k = 0; k < 16; ++k) {
                const float T  = (float)k * (1.0f / 15.0f);
                const float d0 = n0 - T, d1 = n1 - T;
                fr[73 + k] = (_Float16)__expf(-50.0f * d0 * d0);   // 1/(2*sigma^2)=50
                fr[89 + k] = (_Float16)__expf(-50.0f * d1 * d1);
            }
#pragma unroll
            for (int k = 105; k < FEATK; ++k) fr[k] = (_Float16)0.0f;
        }

        // ---- Two 16-point M tiles per wave ----
#pragma unroll
        for (int mt = 0; mt < 2; ++mt) {
            // Layer 1: [16x128] x [128x64]
            const _Float16* arow = &sFeat[w][(mt * 16 + ln) * FEATK];
            half16 a0 = load_A(arow, 0, hs);
            half16 a1 = load_A(arow, 1, hs);
            half16 a2 = load_A(arow, 2, hs);
            half16 a3 = load_A(arow, 3, hs);
#pragma unroll
            for (int nt = 0; nt < 4; ++nt) {
                v8f c = {};
                const _Float16* wrow = &sW1[(nt * 16 + ln) * FEATK + hs * 16];
                c = wmma_f16(a0, load_B(wrow + 0 * 32), c);
                c = wmma_f16(a1, load_B(wrow + 1 * 32), c);
                c = wmma_f16(a2, load_B(wrow + 2 * 32), c);
                c = wmma_f16(a3, load_B(wrow + 3 * 32), c);
                const float bias = sB1[nt * 16 + ln];
#pragma unroll
                for (int r = 0; r < 8; ++r)
                    sAct[w][(r + 8 * hs) * HID + nt * 16 + ln] =
                        (_Float16)__sinf(c[r] + bias);
            }

            // Layer 2: [16x64] x [64x64]
            const _Float16* arow2 = &sAct[w][ln * HID];
            half16 e0 = load_A(arow2, 0, hs);
            half16 e1 = load_A(arow2, 1, hs);
#pragma unroll
            for (int nt = 0; nt < 4; ++nt) {
                v8f c = {};
                const _Float16* wrow = &sW2[(nt * 16 + ln) * HID + hs * 16];
                c = wmma_f16(e0, load_B(wrow + 0), c);
                c = wmma_f16(e1, load_B(wrow + 32), c);
                const float bias = sB2[nt * 16 + ln];
#pragma unroll
                for (int r = 0; r < 8; ++r)
                    sAct[w][(r + 8 * hs) * HID + nt * 16 + ln] =
                        (_Float16)__sinf(c[r] + bias);
            }

            // Layer 3: [16x64] x [64x16] (only n<3 valid)
            const _Float16* arow3 = &sAct[w][ln * HID];
            half16 g0 = load_A(arow3, 0, hs);
            half16 g1 = load_A(arow3, 1, hs);
            v8f c3 = {};
            {
                const _Float16* wrow = &sW3[ln * HID + hs * 16];
                c3 = wmma_f16(g0, load_B(wrow + 0), c3);
                c3 = wmma_f16(g1, load_B(wrow + 32), c3);
            }
            if (ln < 3) {
                const float bb = sB3[ln];
#pragma unroll
                for (int r = 0; r < 8; ++r) {
                    const int p = base + mt * 16 + r + 8 * hs;
                    out[p * 3 + ln] = c3[r] + bb + bases[p * 3 + ln];
                }
            }
        }
    }
}

extern "C" void kernel_launch(void* const* d_in, const int* in_sizes, int n_in,
                              void* d_out, int out_size, void* d_ws, size_t ws_size,
                              hipStream_t stream) {
    (void)in_sizes; (void)n_in; (void)out_size; (void)d_ws; (void)ws_size;
    const float* bases     = (const float*)d_in[0];
    const float* normals   = (const float*)d_in[1];
    const float* encodings = (const float*)d_in[2];
    const float* W1 = (const float*)d_in[3];
    const float* b1 = (const float*)d_in[4];
    const float* W2 = (const float*)d_in[5];
    const float* b2 = (const float*)d_in[6];
    const float* W3 = (const float*)d_in[7];
    const float* b3 = (const float*)d_in[8];
    float* out = (float*)d_out;

    dim3 block(64);          // 2 waves; ~47 KB LDS/block -> several blocks/WGP
    dim3 grid(2048);         // 4096 waves grid-striding over 62500 tiles
    neural_points_kernel<<<grid, block, 0, stream>>>(
        bases, normals, encodings, W1, b1, W2, b2, W3, b3, out);
}
